// AttnNO_32719060861272
// MI455X (gfx1250) — matrix operand order repaired and checked by reading, not compiled, not run on hardware
//
#include <hip/hip_runtime.h>
#include <math.h>

typedef __attribute__((ext_vector_type(16))) _Float16 v16h;
typedef __attribute__((ext_vector_type(8)))  _Float16 v8h;
typedef __attribute__((ext_vector_type(8)))  float    v8f;

#define N_PTS 8192
#define C_DIM 128
#define HEADS 8
#define KNBR  32
#define FC_DIM 256

// ---- CDNA5 async global->LDS path (guarded; falls back to sync copy) ----
#if defined(__HIP_DEVICE_COMPILE__) && __has_builtin(__builtin_amdgcn_global_load_async_to_lds_b128)
#define HAVE_ASYNC_LDS 1
typedef int v4i_vs __attribute__((vector_size(16)));               // int4 (gcc-vector style)
typedef __attribute__((address_space(1))) v4i_vs GInt4;            // global int4
typedef __attribute__((address_space(3))) v4i_vs LInt4;            // LDS int4
#else
#define HAVE_ASYNC_LDS 0
#endif

__device__ __forceinline__ void async_wait_all() {
#if defined(__HIP_DEVICE_COMPILE__)
#if __has_builtin(__builtin_amdgcn_s_wait_asynccnt)
    __builtin_amdgcn_s_wait_asynccnt(0);
#elif HAVE_ASYNC_LDS
    asm volatile("s_wait_asynccnt 0x0" ::: "memory");
#endif
#endif
}

__device__ __forceinline__ float gelu_erf(float x) {
    return 0.5f * x * (1.0f + erff(x * 0.70710678118654752f));
}

// ---- cross-lane reductions (wave32; width-16 variants stay inside the
// ---- half-wave that owns one 16x16 C-tile row group) ----
__device__ __forceinline__ float redmax16(float v) {
    v = fmaxf(v, __shfl_xor(v, 1, 16));
    v = fmaxf(v, __shfl_xor(v, 2, 16));
    v = fmaxf(v, __shfl_xor(v, 4, 16));
    v = fmaxf(v, __shfl_xor(v, 8, 16));
    return v;
}
__device__ __forceinline__ float redsum16(float v) {
    v += __shfl_xor(v, 1, 16);
    v += __shfl_xor(v, 2, 16);
    v += __shfl_xor(v, 4, 16);
    v += __shfl_xor(v, 8, 16);
    return v;
}
__device__ __forceinline__ float redmax32(float v) {
#pragma unroll
    for (int m = 16; m; m >>= 1) v = fmaxf(v, __shfl_xor(v, m, 32));
    return v;
}
__device__ __forceinline__ float redsum32(float v) {
#pragma unroll
    for (int m = 16; m; m >>= 1) v += __shfl_xor(v, m, 32);
    return v;
}

// ---- WMMA A-fragment builders (16x32 f16 tile, M=lane%16).
// Layout: element e (0..15) holds K = 32*step + half16 + e,
// where half16 = 16 for lanes 16..31.
__device__ __forceinline__ v16h a_frag_from_f32(const float* row, int kstep, int lane) {
    const int b0 = kstep * 32 + ((lane & 16) ? 16 : 0);
    v16h a;
#pragma unroll
    for (int e = 0; e < 8; ++e) a[e] = (_Float16)row[b0 + e];
#pragma unroll
    for (int e = 0; e < 8; ++e) a[8 + e] = (_Float16)row[b0 + 8 + e];
    return a;
}
__device__ __forceinline__ v16h a_frag_from_f16(const _Float16* row, int kstep, int lane) {
    const int b0 = kstep * 32 + ((lane & 16) ? 16 : 0);
    v8h lo = *(const v8h*)(row + b0);
    v8h hi = *(const v8h*)(row + b0 + 8);
    v16h a;
#pragma unroll
    for (int e = 0; e < 8; ++e) { a[e] = lo[e]; a[8 + e] = hi[e]; }
    return a;
}

// =====================================================================
// fc0: [N,3] @ [3,128] + b   (tiny K -> plain VALU)
// =====================================================================
__global__ __launch_bounds__(256) void fc0_kernel(
    const float* __restrict__ x, const float* __restrict__ w,
    const float* __restrict__ b, float* __restrict__ out)
{
    const int i = blockIdx.x * 256 + threadIdx.x;     // i < N*128
    const int n = i >> 7, c = i & 127;
    float v = b[c];
    v += x[n * 3 + 0] * w[0 * 128 + c];
    v += x[n * 3 + 1] * w[1 * 128 + c];
    v += x[n * 3 + 2] * w[2 * 128 + c];
    out[i] = v;
}

// =====================================================================
// WMMA GEMM: Y[:,col0:col0+128] = X[N,128] @ W[128,wStride] + bias
// Wt (f16) staged in LDS.  8 waves x 16 rows = 128 rows per block.
// B fragments preloaded per tile so WMMAs issue back-to-back.
// =====================================================================
__global__ __launch_bounds__(256) void gemm128_wmma(
    const float* __restrict__ X, const float* __restrict__ W,
    const float* __restrict__ bias, float* __restrict__ outF32,
    _Float16* __restrict__ outF16, int wStride, int col0, int outStride,
    int applyGelu)
{
    __shared__ _Float16 ldsWt[128 * 128];   // Wt[c][d] = W[d][col0+c]
    const int tid = threadIdx.x;
    for (int idx = tid; idx < 128 * 128; idx += 256) {
        const int c = idx >> 7, d = idx & 127;
        ldsWt[c * 128 + d] = (_Float16)W[d * wStride + col0 + c];
    }
    __syncthreads();

    const int lane = tid & 31, wid = tid >> 5;
    const int rowbase = blockIdx.x * 128 + wid * 16;
    const int mrow = rowbase + (lane & 15);
    const int nloc = lane & 15;
    const int rowoff = (lane & 16) ? 8 : 0;

    v16h a[4];
#pragma unroll
    for (int s = 0; s < 4; ++s) a[s] = a_frag_from_f32(X + mrow * 128, s, lane);

#pragma unroll
    for (int t = 0; t < 8; ++t) {
        v16h b[4];
#pragma unroll
        for (int s = 0; s < 4; ++s) {
            const int koff = s * 32 + ((lane & 16) ? 16 : 0);
            b[s] = *(const v16h*)(&ldsWt[(t * 16 + nloc) * 128 + koff]);
        }
        v8f acc = {};
#pragma unroll
        for (int s = 0; s < 4; ++s)
            acc = __builtin_amdgcn_wmma_f32_16x16x32_f16(false, a[s], false, b[s],
                                                         (short)0, acc, false, false);
        const int col = col0 + t * 16 + nloc;
        const float bv = bias[col];
#pragma unroll
        for (int r = 0; r < 8; ++r) {
            float v = acc[r] + bv;
            if (applyGelu) v = gelu_erf(v);
            const int row = rowbase + r + rowoff;
            if (outF32) outF32[row * outStride + col] = v;
            if (outF16) outF16[row * outStride + col] = (_Float16)v;
        }
    }
}

// =====================================================================
// Flash attention (global):  O = (softmax(Q K^T / sqrt(128)) / N) @ V
// wave -> 16 query rows; block (8 waves) -> 128 rows; 32-key blocks
// double-buffered through LDS (K: ASYNCcnt-tracked async global->LDS
// copy, V: transposed via VGPR scatter).  Online softmax.
// =====================================================================
__global__ __launch_bounds__(256) void flash_attn_wmma(
    const _Float16* __restrict__ Q, const _Float16* __restrict__ Kh,
    const _Float16* __restrict__ V, float* __restrict__ O)
{
    __shared__ _Float16 ldsK[2][32 * 128];     // K block, row-major
    __shared__ _Float16 ldsVt[2][128 * 32];    // V block, transposed [d][key]
    __shared__ _Float16 ldsP[8 * 16 * 32];     // per-wave P re-layout patch

    const int tid = threadIdx.x, lane = tid & 31, wid = tid >> 5;
    const int rowbase = blockIdx.x * 128 + wid * 16;
    const int mrow = rowbase + (lane & 15);
    const int nloc = lane & 15;
    const int rowoff = (lane & 16) ? 8 : 0;

    v16h aq[4];
#pragma unroll
    for (int s = 0; s < 4; ++s) aq[s] = a_frag_from_f16(Q + mrow * 128, s, lane);

    v8f o[8] = {};
    float mr[8], lr[8];
#pragma unroll
    for (int r = 0; r < 8; ++r) { mr[r] = -1e30f; lr[r] = 0.f; }

    // cooperative staging: each thread moves 16 contiguous halves of K and V
    const int base = tid * 16;                 // covers 32*128 halves
    const int kr = base >> 7, kc = base & 127;

    auto stage = [&](int jb, int buf) {
        const _Float16* gK = Kh + (size_t)(jb + kr) * 128 + kc;
#if HAVE_ASYNC_LDS
        __builtin_amdgcn_global_load_async_to_lds_b128((GInt4*)gK,
                                                       (LInt4*)&ldsK[buf][base], 0, 0);
        __builtin_amdgcn_global_load_async_to_lds_b128((GInt4*)(gK + 8),
                                                       (LInt4*)&ldsK[buf][base + 8], 0, 0);
#else
        *(v16h*)(&ldsK[buf][base]) = *(const v16h*)gK;
#endif
        v16h vv = *(const v16h*)(V + (size_t)(jb + kr) * 128 + kc);
#pragma unroll
        for (int e = 0; e < 16; ++e) ldsVt[buf][(kc + e) * 32 + kr] = vv[e];
        if (jb + 32 < N_PTS) {                 // prefetch the block after next
            __builtin_prefetch(Kh + (size_t)(jb + 32 + kr) * 128 + kc, 0, 1);
            __builtin_prefetch(V  + (size_t)(jb + 32 + kr) * 128 + kc, 0, 1);
        }
    };

    stage(0, 0);

    const float scale = 0.08838834764831845f;  // 1/sqrt(128)
    _Float16* myP = &ldsP[wid * 16 * 32];

    for (int jb = 0; jb < N_PTS; jb += 32) {
        const int cur = (jb >> 5) & 1;
        async_wait_all();                      // my async copies for `cur` done
        __syncthreads();                       // everyone's stage + prior reads done
        if (jb + 32 < N_PTS) stage(jb + 32, cur ^ 1);

        // S = Q K^T : preload all 8 B fragments, then interleave 2 chains
        v16h bk0[4], bk1[4];
#pragma unroll
        for (int s = 0; s < 4; ++s) {
            const int koff = s * 32 + ((lane & 16) ? 16 : 0);
            bk0[s] = *(const v16h*)(&ldsK[cur][nloc * 128 + koff]);
            bk1[s] = *(const v16h*)(&ldsK[cur][(16 + nloc) * 128 + koff]);
        }
        v8f s0 = {}, s1 = {};
#pragma unroll
        for (int s = 0; s < 4; ++s) {
            s0 = __builtin_amdgcn_wmma_f32_16x16x32_f16(false, aq[s], false, bk0[s],
                                                        (short)0, s0, false, false);
            s1 = __builtin_amdgcn_wmma_f32_16x16x32_f16(false, aq[s], false, bk1[s],
                                                        (short)0, s1, false, false);
        }

        // online softmax update per row (rows r+rowoff live in this half-wave)
#pragma unroll
        for (int r = 0; r < 8; ++r) {
            const float x0 = s0[r] * scale, x1 = s1[r] * scale;
            const float rmax = redmax16(fmaxf(x0, x1));
            const float mnew = fmaxf(mr[r], rmax);
            const float corr = __expf(mr[r] - mnew);
            const float p0 = __expf(x0 - mnew), p1 = __expf(x1 - mnew);
            lr[r] = lr[r] * corr + redsum16(p0 + p1);
            mr[r] = mnew;
#pragma unroll
            for (int j = 0; j < 8; ++j) o[j][r] *= corr;
            const int prow = r + rowoff;
            myP[prow * 32 + nloc] = (_Float16)p0;        // keys 0..15
            myP[prow * 32 + 16 + nloc] = (_Float16)p1;   // keys 16..31
        }
        // same-wave DS ops are in-order: safe to reload P in A-layout now
        v16h ap = a_frag_from_f16(myP + (lane & 15) * 32, 0, lane);

        // O += P @ V : preload 8 V fragments, then 8 independent WMMAs
        v16h bv[8];
#pragma unroll
        for (int j = 0; j < 8; ++j) {
            const int koff = (lane & 16) ? 16 : 0;
            bv[j] = *(const v16h*)(&ldsVt[cur][(j * 16 + nloc) * 32 + koff]);
        }
#pragma unroll
        for (int j = 0; j < 8; ++j)
            o[j] = __builtin_amdgcn_wmma_f32_16x16x32_f16(false, ap, false, bv[j],
                                                          (short)0, o[j], false, false);
    }

#pragma unroll
    for (int r = 0; r < 8; ++r) {
        const float inv = 1.0f / (lr[r] * (float)N_PTS);  // softmax norm + /N
        const int row = rowbase + r + rowoff;
#pragma unroll
        for (int j = 0; j < 8; ++j)
            O[row * 128 + j * 16 + nloc] = o[j][r] * inv;
    }
}

// =====================================================================
// Local attention: K=32 neighbors == 32 lanes of a wave.
// wave -> one node, loops over 8 heads of dim 16.
// =====================================================================
__global__ __launch_bounds__(256) void local_attn_kernel(
    const _Float16* __restrict__ Q, const _Float16* __restrict__ Kh,
    const _Float16* __restrict__ V, const int* __restrict__ nbr,
    float* __restrict__ O)
{
    const int lane = threadIdx.x & 31, wid = threadIdx.x >> 5;
    const int n = blockIdx.x * 8 + wid;
    const int j = nbr[n * KNBR + lane];
#pragma unroll
    for (int h = 0; h < HEADS; ++h) {
        v16h q = *(const v16h*)(Q + n * 128 + h * 16);
        v16h k = *(const v16h*)(Kh + j * 128 + h * 16);
        float s = 0.f;
#pragma unroll
        for (int e = 0; e < 16; ++e) s += (float)q[e] * (float)k[e];
        s *= 0.25f;                               // 1/sqrt(16)
        const float mx = redmax32(s);
        float p = __expf(s - mx);
        p /= redsum32(p);
        v16h v = *(const v16h*)(V + j * 128 + h * 16);
        float t[16];
#pragma unroll
        for (int e = 0; e < 16; ++e) t[e] = p * (float)v[e];
#pragma unroll
        for (int msk = 16; msk; msk >>= 1) {
#pragma unroll
            for (int e = 0; e < 16; ++e) t[e] += __shfl_xor(t[e], msk, 32);
        }
        if (lane == 0) {
#pragma unroll
            for (int e = 0; e < 16; ++e) O[n * 128 + h * 16 + e] = t[e];
        }
    }
}

// =====================================================================
// y = opt_gelu(A + R)      (residual merge epilogue)
// =====================================================================
__global__ __launch_bounds__(256) void addres_kernel(
    const float* __restrict__ A, const float* __restrict__ R,
    float* __restrict__ out, int applyGelu)
{
    const int i = blockIdx.x * 256 + threadIdx.x;
    float v = A[i] + R[i];
    if (applyGelu) v = gelu_erf(v);
    out[i] = v;
}

// =====================================================================
// fc2: [N,256] @ [256,1] + b  -> one node per wave, lane-strided dot
// =====================================================================
__global__ __launch_bounds__(256) void fc2_kernel(
    const float* __restrict__ F, const float* __restrict__ w,
    const float* __restrict__ b, float* __restrict__ out)
{
    const int lane = threadIdx.x & 31, wid = threadIdx.x >> 5;
    const int n = blockIdx.x * 8 + wid;
    float v = 0.f;
#pragma unroll
    for (int f = 0; f < 256; f += 32) v += F[n * 256 + f + lane] * w[f + lane];
    v = redsum32(v);
    if (lane == 0) out[n] = v + b[0];
}

// =====================================================================
extern "C" void kernel_launch(void* const* d_in, const int* in_sizes, int n_in,
                              void* d_out, int out_size, void* d_ws, size_t ws_size,
                              hipStream_t stream)
{
    (void)in_sizes; (void)n_in; (void)out_size; (void)ws_size;
    const float* x     = (const float*)d_in[0];
    const float* fc0_w = (const float*)d_in[1];
    const float* fc0_b = (const float*)d_in[2];
    const float *qw[3], *qb[3], *kw[3], *kb[3], *vw[3], *vb[3], *ww[3], *wb[3];
    for (int i = 0; i < 3; ++i) {
        const int base = 3 + i * 8;
        qw[i] = (const float*)d_in[base + 0]; qb[i] = (const float*)d_in[base + 1];
        kw[i] = (const float*)d_in[base + 2]; kb[i] = (const float*)d_in[base + 3];
        vw[i] = (const float*)d_in[base + 4]; vb[i] = (const float*)d_in[base + 5];
        ww[i] = (const float*)d_in[base + 6]; wb[i] = (const float*)d_in[base + 7];
    }
    const float* fc1_w = (const float*)d_in[27];
    const float* fc1_b = (const float*)d_in[28];
    const float* fc2_w = (const float*)d_in[29];
    const float* fc2_b = (const float*)d_in[30];
    const int*   nbr   = (const int*)d_in[31];
    float* out = (float*)d_out;

    char* ws = (char*)d_ws;
    float*    bufA = (float*)ws;     ws += (size_t)N_PTS * 128 * 4;
    float*    bufB = (float*)ws;     ws += (size_t)N_PTS * 128 * 4;
    float*    bufR = (float*)ws;     ws += (size_t)N_PTS * 128 * 4;
    _Float16* Qh   = (_Float16*)ws;  ws += (size_t)N_PTS * 128 * 2;
    _Float16* Kh   = (_Float16*)ws;  ws += (size_t)N_PTS * 128 * 2;
    _Float16* Vh   = (_Float16*)ws;  ws += (size_t)N_PTS * 128 * 2;
    float*    Fbuf = (float*)ws;     ws += (size_t)N_PTS * 256 * 4;

    const dim3 blk(256);
    const int gElem = N_PTS * 128 / 256;   // elementwise grid
    const int gRows = N_PTS / 128;         // 128-row tile grid

    auto proj = [&](const float* X, const float* W, const float* B,
                    float* oF32, _Float16* oF16) {
        gemm128_wmma<<<gRows, blk, 0, stream>>>(X, W, B, oF32, oF16, 128, 0, 128, 0);
    };

    // fc0
    fc0_kernel<<<gElem, blk, 0, stream>>>(x, fc0_w, fc0_b, bufA);

    // ---- layer 0: global attention + residual proj, GELU ----
    proj(bufA, qw[0], qb[0], nullptr, Qh);
    proj(bufA, kw[0], kb[0], nullptr, Kh);
    proj(bufA, vw[0], vb[0], nullptr, Vh);
    proj(bufA, ww[0], wb[0], bufR, nullptr);
    flash_attn_wmma<<<gRows, blk, 0, stream>>>(Qh, Kh, Vh, bufB);
    addres_kernel<<<gElem, blk, 0, stream>>>(bufB, bufR, bufB, 1);   // h1 = bufB

    // ---- layer 1: local attention + residual proj, GELU ----
    proj(bufB, qw[1], qb[1], nullptr, Qh);
    proj(bufB, kw[1], kb[1], nullptr, Kh);
    proj(bufB, vw[1], vb[1], nullptr, Vh);
    proj(bufB, ww[1], wb[1], bufR, nullptr);
    local_attn_kernel<<<N_PTS / 8, blk, 0, stream>>>(Qh, Kh, Vh, nbr, bufA);
    addres_kernel<<<gElem, blk, 0, stream>>>(bufA, bufR, bufA, 1);   // h2 = bufA

    // ---- layer 2: global attention + residual proj, no activation ----
    proj(bufA, qw[2], qb[2], nullptr, Qh);
    proj(bufA, kw[2], kb[2], nullptr, Kh);
    proj(bufA, vw[2], vb[2], nullptr, Vh);
    proj(bufA, ww[2], wb[2], bufR, nullptr);
    flash_attn_wmma<<<gRows, blk, 0, stream>>>(Qh, Kh, Vh, bufB);
    addres_kernel<<<gElem, blk, 0, stream>>>(bufB, bufR, bufB, 0);   // h3 = bufB

    // ---- fc1 (128->256, GELU): two 128-column slabs ----
    gemm128_wmma<<<gRows, blk, 0, stream>>>(bufB, fc1_w, fc1_b, Fbuf, nullptr,
                                            FC_DIM, 0,   FC_DIM, 1);
    gemm128_wmma<<<gRows, blk, 0, stream>>>(bufB, fc1_w, fc1_b, Fbuf, nullptr,
                                            FC_DIM, 128, FC_DIM, 1);

    // ---- fc2 (256->1) ----
    fc2_kernel<<<N_PTS / 8, blk, 0, stream>>>(Fbuf, fc2_w, fc2_b, out);
}
